// SimpleS4Wrapper_53360673685992
// MI455X (gfx1250) — matrix operand order repaired and checked by reading, not compiled, not run on hardware
//
#include <hip/hip_runtime.h>
#include <math.h>

// Problem constants (B, H, L, CH=1, D2) from the reference.
#define B_  8
#define H_  512
#define L_  2048
#define D2_ 32
#define FRPAD 64
#define FRLEN (L_ + FRPAD)            // reversed+padded f row length (elems)

typedef __attribute__((ext_vector_type(16))) __bf16 v16bf;
typedef __attribute__((ext_vector_type(8)))  float  v8f;
typedef __attribute__((ext_vector_type(4)))  unsigned int v4u;
typedef __attribute__((ext_vector_type(8)))  int  v8i;
typedef __attribute__((ext_vector_type(4)))  int  v4i;

// LDS byte offset of a __shared__ object: generic shared addr low 32 bits are
// the LDS offset (ISA: LDS aperture keeps offset in ADDR[31:0]).
__device__ inline unsigned lds_off(const void* p) {
  return (unsigned)(unsigned long long)(uintptr_t)p;
}

// ---------------------------------------------------------------------------
// Tensor Data Mover: 1-D row load Global(bf16) -> LDS.
// D# per CDNA5 ISA:
//   group0: count=1 | lds_addr | global_addr[56:0] | type=2
//   group1: data_size=2B, tensor_dim0 = tile_dim0 = nelem, stride0 = nelem
// ---------------------------------------------------------------------------
__device__ inline void tdm_load_row_bf16(const __bf16* gsrc, unsigned ldsbyte,
                                         int nelem) {
  unsigned long long ga = (unsigned long long)(uintptr_t)gsrc;
  v4u g0;
  g0[0] = 1u;                                   // count=1, user descriptor
  g0[1] = ldsbyte;                              // lds_addr (bytes)
  g0[2] = (unsigned)(ga & 0xffffffffu);         // global_addr[31:0]
  g0[3] = (unsigned)((ga >> 32) & 0x01ffffffu)  // global_addr[56:32]
          | (2u << 30);                         // type=2 ("image")
  v8i g1;
  g1[0] = 0x10000;                              // wg_mask=0, data_size=1 (2B)
  g1[1] = (int)((nelem & 0xffff) << 16);        // tensor_dim0[15:0] @ bits63:48
  g1[2] = (int)((nelem >> 16) & 0xffff);        // tensor_dim0[31:16]; dim1=0
  g1[3] = (int)((nelem & 0xffff) << 16);        // tile_dim0 @ bits127:112
  g1[4] = 0;                                    // tile_dim1/2 unused
  g1[5] = nelem;                                // tensor_dim0_stride[31:0]
  g1[6] = 0;
  g1[7] = 0;
  v4i z = {0, 0, 0, 0};
#if defined(__clang_major__) && (__clang_major__ >= 23)
  v8i z8 = {0, 0, 0, 0, 0, 0, 0, 0};
  __builtin_amdgcn_tensor_load_to_lds(g0, g1, z, z, z8, 0);
#else
  __builtin_amdgcn_tensor_load_to_lds(g0, g1, z, z, 0);
#endif
}

// ---------------------------------------------------------------------------
// Kernel 1: basis synthesis.
//   f[h,l]  = 2T * sum_d (b*c)_d  * exp(-|a|_d z_l) * cos(theta_d z_l)
//   k0[h,l] = 2  * sum_d (c*x0)_d * exp(-|a|_d z_l) * cos(theta_d z_l)
// Emits f in f32 (epilogue corrections) and a REVERSED, zero-padded bf16
// image fr[h][i] = f[h, L-1-i] so the conv A-tile gather is branchless:
// causality (negative lag) lands in the zero pad instead of a compare.
// ---------------------------------------------------------------------------
__global__ __launch_bounds__(256) void s4_basis_kernel(
    const float* __restrict__ a,  const float* __restrict__ th,
    const float* __restrict__ bb, const float* __restrict__ cc,
    const float* __restrict__ x0,
    float* __restrict__ f32f, float* __restrict__ k0f,
    __bf16* __restrict__ frv)
{
  int idx = blockIdx.x * 256 + threadIdx.x;     // over H_*L_
  int h = idx >> 11;                            // / L_ (L_=2048)
  int l = idx & (L_ - 1);
  const float T = 1.0f / (float)(L_ - 1);
  float zl = T * (float)l;
  const float* ah  = a  + h * D2_;
  const float* thh = th + h * D2_;
  const float* bh  = bb + h * D2_;
  const float* ch  = cc + h * D2_;
  const float* xh  = x0 + h * D2_;
  float s = 0.0f, s0 = 0.0f;
#pragma unroll 4
  for (int d = 0; d < D2_; ++d) {
    float e = __expf(-fabsf(ah[d]) * zl) * __cosf(thh[d] * zl);
    s  += bh[d] * ch[d] * e;
    s0 += ch[d] * xh[d] * e;
  }
  float fv = 2.0f * T * s;
  f32f[idx] = fv;
  k0f[idx]  = 2.0f * s0;
  __bf16* frh = frv + (size_t)h * FRLEN;
  frh[L_ - 1 - l] = (__bf16)fv;                 // reversed image
  if (l < FRPAD) frh[L_ + l] = (__bf16)0.0f;    // causality pad (zeros)
}

// ---------------------------------------------------------------------------
// Kernel 2: causal Toeplitz convolution via v_wmma_f32_16x16x32_bf16.
// One wave owns a 16-wide l-tile of one h:  Y[r,n] = sum_m F[r,m] U[m,n].
//   A[r,K] = f[l0+r-(m0+K)] = fr_lds[(L-1-l0-r+m0) + K]   (branchless)
//   B[K,n] = u_lds[n][m0+K], row 8 is all-zero for the N-pad columns.
// f row arrives in LDS via the Tensor Data Mover (wave 0, TENSORcnt).
// Inner loop is software-pipelined (load kk+1, WMMA kk) with a scalarized
// (readfirstlane) trip count so the compiler emits an s_cbranch loop and the
// ds-wait for the next tiles overlaps the current XDL op.
// ---------------------------------------------------------------------------
__global__ __launch_bounds__(256) void s4_conv_kernel(
    const float* __restrict__ u,
    const float* __restrict__ f32f, const __bf16* __restrict__ frv,
    const float* __restrict__ k0f,  const float* __restrict__ Dv,
    __bf16* __restrict__ g)
{
  __shared__ __bf16 frl[FRLEN];                 // 4.2 KB reversed f row
  __shared__ __bf16 ubl[B_ + 1][L_];            // 36 KB; row B_ = zeros

  const int tid  = threadIdx.x;
  const int wave = tid >> 5;
  const int lane = tid & 31;
  const int h    = blockIdx.y;

  // --- TDM stage of the reversed f row (one instruction, wave 0) ---
  if (wave == 0) {
    tdm_load_row_bf16(frv + (size_t)h * FRLEN, lds_off(frl), FRLEN);
    __builtin_amdgcn_s_wait_tensorcnt(0);
  }
  // --- cooperative u panel stage, f32 -> bf16 (+ explicit zero row) ---
  for (int i = tid; i < (B_ + 1) * L_; i += 256) {
    int b = i >> 11, l = i & (L_ - 1);
    float uv = (b < B_) ? u[((size_t)b * H_ + h) * L_ + l] : 0.0f;
    ubl[b][l] = (__bf16)uv;
  }
  __syncthreads();

  const int ltile = blockIdx.x * 8 + wave;      // 0..127
  const int l0    = ltile * 16;
  const int r     = lane & 15;                  // A row (M) == B column (N)
  const int br    = (r < B_) ? r : B_;          // zero row for pad columns
  const int baseKA = (lane < 16) ? 0 : 8;       // ISA 16-bit A lane layout
  const int baseKB = (lane < 16) ? 0 : 8;       // ISA 16-bit B lane layout
  const int ibase0 = (L_ - 1) - l0 - r + baseKA;

  // wave-uniform trip count -> SGPR so the loop uses s_cbranch, not EXEC ops
  const int nkk = __builtin_amdgcn_readfirstlane(((l0 + 15) >> 5) + 1);

  v8f acc = {};
  v16bf A0, B0;
  {
    const __bf16* fp = &frl[ibase0];
    const __bf16* up = &ubl[br][baseKB];
#pragma unroll
    for (int j = 0; j < 8; ++j) { A0[j] = fp[j]; A0[j + 8] = fp[16 + j]; }
#pragma unroll
    for (int j = 0; j < 8; ++j) { B0[j] = up[j]; B0[j + 8] = up[16 + j]; }
  }
  for (int kk = 1; kk < nkk; ++kk) {
    // issue next tiles' LDS loads before consuming the current ones
    const __bf16* fp = &frl[ibase0 + (kk << 5)];
    const __bf16* up = &ubl[br][(kk << 5) + baseKB];
    v16bf A1, B1;
#pragma unroll
    for (int j = 0; j < 8; ++j) { A1[j] = fp[j]; A1[j + 8] = fp[16 + j]; }
#pragma unroll
    for (int j = 0; j < 8; ++j) { B1[j] = up[j]; B1[j + 8] = up[16 + j]; }
    acc = __builtin_amdgcn_wmma_f32_16x16x32_bf16(
        false, A0, false, B0, (short)0, acc, false, false);
    A0 = A1;
    B0 = B1;
  }
  acc = __builtin_amdgcn_wmma_f32_16x16x32_bf16(
      false, A0, false, B0, (short)0, acc, false, false);

  // ---- fused epilogue: corrections + skip + GELU, store bf16 [b][l][h] ----
  if (r < B_) {
    const size_t ubase = ((size_t)r * H_ + h) * L_;
    const float u0 = u[ubase];                  // u[b,h,0]
    const float f0 = f32f[(size_t)h * L_];      // f[h,0]
    const float Dh = Dv[h];
#pragma unroll
    for (int v = 0; v < 8; ++v) {
      int M = (lane < 16) ? v : v + 8;          // ISA f32 C/D lane layout
      int l = l0 + M;
      float ul = u[ubase + l];
      float y = acc[v];
      y += k0f[(size_t)h * L_ + l];
      y -= 0.5f * (f0 * ul + f32f[(size_t)h * L_ + l] * u0);
      y += Dh * ul;
      float ge = 0.5f * y * (1.0f + erff(y * 0.70710678118f)); // exact GELU
      g[(((size_t)r * L_) + l) * H_ + h] = (__bf16)ge;
    }
  }
}

// ---------------------------------------------------------------------------
// Kernel 3: head GEMM  out[b,h,l] = sum_f W[h,f] * g[b,l,f] + bias[h].
// Block = 8 waves: one 16-row h-tile of W staged in LDS (16 KB bf16), shared
// by 8 waves each owning a 16-wide l-tile. K=512 -> 16 WMMA per tile,
// software-pipelined so global/LDS latency overlaps the XDL ops.
// ---------------------------------------------------------------------------
__global__ __launch_bounds__(256) void s4_head_kernel(
    const float* __restrict__ W, const float* __restrict__ bias,
    const __bf16* __restrict__ g, float* __restrict__ out)
{
  __shared__ __bf16 Wlds[16 * H_];              // 16 KB

  const int tid = threadIdx.x;
  const int h0  = blockIdx.y * 16;
  for (int i = tid; i < 16 * H_; i += 256) {
    int rr = i >> 9, k = i & (H_ - 1);
    Wlds[i] = (__bf16)W[(h0 + rr) * H_ + k];
  }
  __syncthreads();

  const int wave = tid >> 5;
  const int lane = tid & 31;
  const int b    = blockIdx.z;
  const int l0   = (blockIdx.x * 8 + wave) * 16;
  const int n    = lane & 15;
  const int baseKA = (lane < 16) ? 0 : 8;
  const int baseKB = (lane < 16) ? 0 : 8;
  const __bf16* grow = g + ((size_t)b * L_ + (l0 + n)) * H_ + baseKB;
  const __bf16* wrow = &Wlds[n * H_ + baseKA];

  v8f acc = {};
  v16bf A0, B0;
#pragma unroll
  for (int j = 0; j < 8; ++j) { A0[j] = wrow[j]; A0[j + 8] = wrow[16 + j]; }
#pragma unroll
  for (int j = 0; j < 8; ++j) { B0[j] = grow[j]; B0[j + 8] = grow[16 + j]; }

#pragma unroll 4
  for (int kk = 1; kk < H_ / 32; ++kk) {
    const __bf16* gp = grow + kk * 32;
    const __bf16* wp = wrow + kk * 32;
    __builtin_prefetch(gp + 128, 0, 0);         // global_prefetch hint
    v16bf A1, B1;
#pragma unroll
    for (int j = 0; j < 8; ++j) { A1[j] = wp[j]; A1[j + 8] = wp[16 + j]; }
#pragma unroll
    for (int j = 0; j < 8; ++j) { B1[j] = gp[j]; B1[j + 8] = gp[16 + j]; }
    acc = __builtin_amdgcn_wmma_f32_16x16x32_bf16(
        false, A0, false, B0, (short)0, acc, false, false);
    A0 = A1;
    B0 = B1;
  }
  acc = __builtin_amdgcn_wmma_f32_16x16x32_bf16(
      false, A0, false, B0, (short)0, acc, false, false);

#pragma unroll
  for (int v = 0; v < 8; ++v) {
    int M = (lane < 16) ? v : v + 8;
    int h = h0 + M;
    out[((size_t)b * H_ + h) * L_ + l0 + n] = acc[v] + bias[h];
  }
}

// ---------------------------------------------------------------------------
extern "C" void kernel_launch(void* const* d_in, const int* in_sizes, int n_in,
                              void* d_out, int out_size, void* d_ws, size_t ws_size,
                              hipStream_t stream) {
  const float* u     = (const float*)d_in[0];   // (B,H,L)
  const float* a     = (const float*)d_in[1];   // (1,H,32)
  const float* theta = (const float*)d_in[2];
  const float* bco   = (const float*)d_in[3];
  const float* cco   = (const float*)d_in[4];
  const float* x0    = (const float*)d_in[5];
  const float* Dv    = (const float*)d_in[6];   // (1,H)
  const float* W     = (const float*)d_in[7];   // (H, H)
  const float* bias  = (const float*)d_in[8];   // (H,)
  float* out = (float*)d_out;

  // workspace carve-up (~29 MB)
  char* ws = (char*)d_ws;
  float*  f32f = (float*)ws;                                   // H*L f32
  float*  k0f  = (float*)(ws + (size_t)4 * H_ * L_);           // H*L f32
  __bf16* frv  = (__bf16*)(ws + (size_t)8 * H_ * L_);          // H*FRLEN bf16
  __bf16* g    = (__bf16*)(ws + (size_t)12 * H_ * L_);         // B*L*H bf16

  s4_basis_kernel<<<(H_ * L_) / 256, 256, 0, stream>>>(
      a, theta, bco, cco, x0, f32f, k0f, frv);

  s4_conv_kernel<<<dim3(L_ / (16 * 8), H_, 1), 256, 0, stream>>>(
      u, f32f, frv, k0f, Dv, g);

  s4_head_kernel<<<dim3(L_ / (16 * 8), H_ / 16, B_), 256, 0, stream>>>(
      W, bias, g, out);
}
